// MoleculeGNS_40604620816457
// MI455X (gfx1250) — matrix operand order repaired and checked by reading, not compiled, not run on hardware
//
#include <hip/hip_runtime.h>

typedef __attribute__((ext_vector_type(16))) __bf16 v16bf;
typedef __attribute__((ext_vector_type(8)))  float  v8f;

#define NN 50000
#define EE 500000

// ---------------------------------------------------------------- helpers

union FragU { v16bf v; uint4 q[2]; };

__device__ __forceinline__ v16bf ldfrag(const __bf16* p0, const __bf16* p1) {
  FragU f;
  f.q[0] = *(const uint4*)p0;
  f.q[1] = *(const uint4*)p1;
  return f.v;
}

__device__ __forceinline__ unsigned enc_f(float f) {
  unsigned u = __float_as_uint(f);
  return (u & 0x80000000u) ? ~u : (u | 0x80000000u);
}
__device__ __forceinline__ float dec_f(unsigned u) {
  return (u & 0x80000000u) ? __uint_as_float(u & 0x7FFFFFFFu) : __uint_as_float(~u);
}

// pack two f32 into bf16x2 (round-to-nearest-even), elem a -> low 16
__device__ __forceinline__ unsigned pack_bf16x2(float a, float b) {
  unsigned ua = __float_as_uint(a);
  unsigned ub = __float_as_uint(b);
  ua += 0x7FFFu + ((ua >> 16) & 1u);
  ub += 0x7FFFu + ((ub >> 16) & 1u);
  return (ua >> 16) | (ub & 0xFFFF0000u);
}

__device__ __forceinline__ float silu(float x) {
  return x * __builtin_amdgcn_rcpf(1.0f + __expf(-x));
}

// One 16-row tile hidden layer: dst = silu(src @ W + b), dst pitch fixed 256.
// src: LDS bf16 [16][kp]; wt: global bf16 [nn][kp] (transposed, K-contiguous).
__device__ __forceinline__ void mlp_hidden(const __bf16* src, int kp,
                                           const __bf16* wt, const float* bias,
                                           int nn, __bf16* dst, int lane) {
  const int l16 = lane & 15;
  const int hi  = lane >> 4;
  const __bf16* arow = src + l16 * kp + hi * 8;
  for (int ct = 0; ct < nn; ct += 16) {
    v8f acc = {};
    const __bf16* brow = wt + (size_t)(ct + l16) * kp + hi * 8;
    for (int ks = 0; ks < kp; ks += 32) {
      v16bf a = ldfrag(arow + ks, arow + ks + 16);
      v16bf b = ldfrag(brow + ks, brow + ks + 16);
      acc = __builtin_amdgcn_wmma_f32_16x16x32_bf16(false, a, false, b,
                                                    (short)0, acc, false, false);
    }
    float bv = bias[ct + l16];
#pragma unroll
    for (int i = 0; i < 8; ++i)
      dst[(i + hi * 8) * 256 + ct + l16] = (__bf16)silu(acc[i] + bv);
  }
}

// Final linear layer to f32 LDS, N fixed 128, no activation.
__device__ __forceinline__ void mlp_final(const __bf16* src, int kp,
                                          const __bf16* wt, const float* bias,
                                          float* F, int lane) {
  const int l16 = lane & 15;
  const int hi  = lane >> 4;
  const __bf16* arow = src + l16 * kp + hi * 8;
  for (int ct = 0; ct < 128; ct += 16) {
    v8f acc = {};
    const __bf16* brow = wt + (size_t)(ct + l16) * kp + hi * 8;
    for (int ks = 0; ks < kp; ks += 32) {
      v16bf a = ldfrag(arow + ks, arow + ks + 16);
      v16bf b = ldfrag(brow + ks, brow + ks + 16);
      acc = __builtin_amdgcn_wmma_f32_16x16x32_bf16(false, a, false, b,
                                                    (short)0, acc, false, false);
    }
    float bv = bias[ct + l16];
#pragma unroll
    for (int i = 0; i < 8; ++i)
      F[(i + hi * 8) * 128 + ct + l16] = acc[i] + bv;
  }
}

// ---------------------------------------------------------------- kernels

// f32 [K][N] -> bf16 [N][Kpad] transpose + pad
__global__ void k_convert(const float* __restrict__ src, __bf16* __restrict__ dst,
                          int K, int N, int Kpad) {
  int i = blockIdx.x * blockDim.x + threadIdx.x;
  if (i >= N * Kpad) return;
  int n = i / Kpad, k = i % Kpad;
  dst[i] = (k < K) ? (__bf16)src[(size_t)k * N + n] : (__bf16)0.0f;
}

__global__ void k_fill(float* p, float v, int n) {
  int i = blockIdx.x * blockDim.x + threadIdx.x;
  if (i < n) p[i] = v;
}

// node encoder: embedding gather -> MLP(128,256,256,128)+SiLU -> LN
__global__ void k_node_encoder(const int* __restrict__ an, const float* __restrict__ emb,
                               const __bf16* w0, const float* b0,
                               const __bf16* w1, const float* b1,
                               const __bf16* w2, const float* b2,
                               const float* g, const float* be, float* nodes) {
  __shared__ alignas(16) __bf16 IN[16 * 256];
  __shared__ alignas(16) __bf16 H1[16 * 256];
  __shared__ alignas(16) float  F [16 * 128];
  const int lane = threadIdx.x, tile = blockIdx.x;
  const int row = lane >> 1, half = (lane & 1) * 64;
  const int nid = tile * 16 + row;
  {
    const float2* src = (const float2*)(emb + (size_t)an[nid] * 128 + half);
    unsigned* dst2 = (unsigned*)&IN[row * 128 + half];
    for (int c = 0; c < 32; ++c) { float2 v = src[c]; dst2[c] = pack_bf16x2(v.x, v.y); }
  }
  __syncthreads();
  mlp_hidden(IN, 128, w0, b0, 256, H1, lane);  __syncthreads();
  mlp_hidden(H1, 256, w1, b1, 256, IN, lane);  __syncthreads();
  mlp_final (IN, 256, w2, b2, F, lane);        __syncthreads();
  float s = 0.f, ss = 0.f;
  for (int c = 0; c < 64; ++c) { float v = F[row * 128 + half + c]; s += v; ss += v * v; }
  s  += __shfl_xor(s, 1, 32);
  ss += __shfl_xor(ss, 1, 32);
  float mu = s * (1.f / 128.f), var = ss * (1.f / 128.f) - mu * mu;
  float rs = rsqrtf(var + 1e-5f);
  float* dst = nodes + (size_t)nid * 128 + half;
  for (int c = 0; c < 64; ++c)
    dst[c] = (F[row * 128 + half + c] - mu) * rs * g[half + c] + be[half + c];
}

// edge encoder: raw 23-dim feats -> MLP(23,256,256,128)+SiLU -> LN
// fused: compute layer-0 attention logits on the produced edge row + segment max
__global__ void k_edge_encoder(const float* __restrict__ ef,
                               const __bf16* w0, const float* b0,
                               const __bf16* w1, const float* b1,
                               const __bf16* w2, const float* b2,
                               const float* g, const float* be, float* edges,
                               const float* __restrict__ wrn, const float* __restrict__ brn,
                               const float* __restrict__ wsn, const float* __restrict__ bsn,
                               const int* __restrict__ recv, const int* __restrict__ send,
                               float* rlog, float* slog, unsigned* rmax, unsigned* smax) {
  __shared__ alignas(16) __bf16 IN[16 * 256];
  __shared__ alignas(16) __bf16 H1[16 * 256];
  __shared__ alignas(16) float  F [16 * 128];
  const int lane = threadIdx.x, tile = blockIdx.x;
  if (lane < 16) {
    const float* src = ef + (size_t)(tile * 16 + lane) * 23;
    for (int c = 0; c < 23; ++c) IN[lane * 32 + c] = (__bf16)src[c];
    for (int c = 23; c < 32; ++c) IN[lane * 32 + c] = (__bf16)0.0f;
  }
  __syncthreads();
  mlp_hidden(IN, 32,  w0, b0, 256, H1, lane);  __syncthreads();
  mlp_hidden(H1, 256, w1, b1, 256, IN, lane);  __syncthreads();
  mlp_final (IN, 256, w2, b2, F, lane);        __syncthreads();
  const int row = lane >> 1, half = (lane & 1) * 64;
  const int e = tile * 16 + row;
  float s = 0.f, ss = 0.f;
  for (int c = 0; c < 64; ++c) { float v = F[row * 128 + half + c]; s += v; ss += v * v; }
  s  += __shfl_xor(s, 1, 32);
  ss += __shfl_xor(ss, 1, 32);
  float mu = s * (1.f / 128.f), var = ss * (1.f / 128.f) - mu * mu;
  float rs = rsqrtf(var + 1e-5f);
  float* dst = edges + (size_t)e * 128 + half;
  float dotR = 0.f, dotS = 0.f;
  for (int c = 0; c < 64; ++c) {
    float en = (F[row * 128 + half + c] - mu) * rs * g[half + c] + be[half + c];
    dst[c] = en;
    dotR += en * wrn[half + c];
    dotS += en * wsn[half + c];
  }
  dotR += __shfl_xor(dotR, 1, 32);
  dotS += __shfl_xor(dotS, 1, 32);
  if ((lane & 1) == 0) {
    dotR += brn[0]; dotS += bsn[0];
    rlog[e] = dotR; slog[e] = dotS;
    atomicMax(&rmax[recv[e]], enc_f(dotR));
    atomicMax(&smax[send[e]], enc_f(dotS));
  }
}

// e = exp(logit - max), segment sums
__global__ void k_expsum(const int* __restrict__ recv, const int* __restrict__ send,
                         const unsigned* __restrict__ rmax, const unsigned* __restrict__ smax,
                         float* rw, float* sw, float* rsum, float* ssum, int E) {
  int i = blockIdx.x * blockDim.x + threadIdx.x;
  if (i >= E) return;
  int r = recv[i], s = send[i];
  float mr = dec_f(rmax[r]); if (!(mr > -3.0e38f && mr < 3.0e38f)) mr = 0.f;
  float ms = dec_f(smax[s]); if (!(ms > -3.0e38f && ms < 3.0e38f)) ms = 0.f;
  float er = __expf(rw[i] - mr);
  float es = __expf(sw[i] - ms);
  rw[i] = er; sw[i] = es;
  atomicAdd(&rsum[r], er);
  atomicAdd(&ssum[s], es);
}

// edge interaction: gather [edge|sent|recv] -> MLP(384,256,256,128)+LN,
// edges += upd, attention-weighted atomic scatter, fused next-layer logits
__global__ void k_edge_layer(float* edges, const float* __restrict__ nodes,
                             const int* __restrict__ send, const int* __restrict__ recv,
                             const __bf16* w0, const float* b0,
                             const __bf16* w1, const float* b1,
                             const __bf16* w2, const float* b2,
                             const float* g, const float* be,
                             const float* __restrict__ rw, const float* __restrict__ rsum,
                             const float* __restrict__ sw, const float* __restrict__ ssum,
                             float* ragg, float* sagg,
                             const float* __restrict__ wrn, const float* __restrict__ brn,
                             const float* __restrict__ wsn, const float* __restrict__ bsn,
                             float* nrlog, float* nslog, unsigned* nrmax, unsigned* nsmax) {
  __shared__ alignas(16) __bf16 IN[16 * 384];
  __shared__ alignas(16) __bf16 H1[16 * 256];
  __shared__ alignas(16) float  F [16 * 128];
  __shared__ int   S[16], R[16];
  __shared__ float RW[16], SW[16];
  const int lane = threadIdx.x, tile = blockIdx.x;
  if (lane < 16) {
    int e = tile * 16 + lane;
    int s = send[e], r = recv[e];
    S[lane] = s; R[lane] = r;
    RW[lane] = rw[e] / (rsum[r] + 1e-16f);
    SW[lane] = sw[e] / (ssum[s] + 1e-16f);
  }
  __syncthreads();
  const int row = lane >> 1, half = (lane & 1) * 64;
  const int e = tile * 16 + row;
  {
    const float2* pe = (const float2*)(edges + (size_t)e * 128 + half);
    const float2* ps = (const float2*)(nodes + (size_t)S[row] * 128 + half);
    const float2* pr = (const float2*)(nodes + (size_t)R[row] * 128 + half);
    unsigned* d0 = (unsigned*)&IN[row * 384 +       half];
    unsigned* d1 = (unsigned*)&IN[row * 384 + 128 + half];
    unsigned* d2 = (unsigned*)&IN[row * 384 + 256 + half];
    for (int c = 0; c < 32; ++c) {
      float2 a = pe[c], b = ps[c], d = pr[c];
      d0[c] = pack_bf16x2(a.x, a.y);
      d1[c] = pack_bf16x2(b.x, b.y);
      d2[c] = pack_bf16x2(d.x, d.y);
    }
  }
  __syncthreads();
  mlp_hidden(IN, 384, w0, b0, 256, H1, lane);  __syncthreads();
  mlp_hidden(H1, 256, w1, b1, 256, IN, lane);  __syncthreads();
  mlp_final (IN, 256, w2, b2, F, lane);        __syncthreads();
  float s = 0.f, ss = 0.f;
  for (int c = 0; c < 64; ++c) { float v = F[row * 128 + half + c]; s += v; ss += v * v; }
  s  += __shfl_xor(s, 1, 32);
  ss += __shfl_xor(ss, 1, 32);
  float mu = s * (1.f / 128.f), var = ss * (1.f / 128.f) - mu * mu;
  float rs = rsqrtf(var + 1e-5f);
  const float rwv = RW[row], swv = SW[row];
  float* pe  = edges + (size_t)e * 128 + half;
  float* pra = ragg + (size_t)R[row] * 128 + half;
  float* psa = sagg + (size_t)S[row] * 128 + half;
  const bool do_attn = (wrn != nullptr);
  float dotR = 0.f, dotS = 0.f;
  for (int c = 0; c < 64; ++c) {
    float y = (F[row * 128 + half + c] - mu) * rs * g[half + c] + be[half + c];
    float en = pe[c] + y;             // edges = edges + upd_e (exclusive row)
    pe[c] = en;
    atomicAdd(&pra[c], y * rwv);      // recv_agg += upd_e * recv_attn
    atomicAdd(&psa[c], y * swv);      // sent_agg += upd_e * send_attn
    if (do_attn) { dotR += en * wrn[half + c]; dotS += en * wsn[half + c]; }
  }
  if (do_attn) {
    dotR += __shfl_xor(dotR, 1, 32);
    dotS += __shfl_xor(dotS, 1, 32);
    if ((lane & 1) == 0) {
      dotR += brn[0]; dotS += bsn[0];
      nrlog[e] = dotR; nslog[e] = dotS;
      atomicMax(&nrmax[R[row]], enc_f(dotR));
      atomicMax(&nsmax[S[row]], enc_f(dotS));
    }
  }
}

// node interaction: [nodes|recv_agg|sent_agg] -> MLP+LN, nodes += upd
__global__ void k_node_layer(float* nodes, const float* __restrict__ ragg,
                             const float* __restrict__ sagg,
                             const __bf16* w0, const float* b0,
                             const __bf16* w1, const float* b1,
                             const __bf16* w2, const float* b2,
                             const float* g, const float* be) {
  __shared__ alignas(16) __bf16 IN[16 * 384];
  __shared__ alignas(16) __bf16 H1[16 * 256];
  __shared__ alignas(16) float  F [16 * 128];
  const int lane = threadIdx.x, tile = blockIdx.x;
  const int row = lane >> 1, half = (lane & 1) * 64;
  const size_t n0 = (size_t)(tile * 16 + row) * 128 + half;
  {
    const float2* p0 = (const float2*)(nodes + n0);
    const float2* p1 = (const float2*)(ragg + n0);
    const float2* p2 = (const float2*)(sagg + n0);
    unsigned* d0 = (unsigned*)&IN[row * 384 +       half];
    unsigned* d1 = (unsigned*)&IN[row * 384 + 128 + half];
    unsigned* d2 = (unsigned*)&IN[row * 384 + 256 + half];
    for (int c = 0; c < 32; ++c) {
      float2 a = p0[c], b = p1[c], d = p2[c];
      d0[c] = pack_bf16x2(a.x, a.y);
      d1[c] = pack_bf16x2(b.x, b.y);
      d2[c] = pack_bf16x2(d.x, d.y);
    }
  }
  __syncthreads();
  mlp_hidden(IN, 384, w0, b0, 256, H1, lane);  __syncthreads();
  mlp_hidden(H1, 256, w1, b1, 256, IN, lane);  __syncthreads();
  mlp_final (IN, 256, w2, b2, F, lane);        __syncthreads();
  float s = 0.f, ss = 0.f;
  for (int c = 0; c < 64; ++c) { float v = F[row * 128 + half + c]; s += v; ss += v * v; }
  s  += __shfl_xor(s, 1, 32);
  ss += __shfl_xor(ss, 1, 32);
  float mu = s * (1.f / 128.f), var = ss * (1.f / 128.f) - mu * mu;
  float rs = rsqrtf(var + 1e-5f);
  for (int c = 0; c < 64; ++c) {
    float y = (F[row * 128 + half + c] - mu) * rs * g[half + c] + be[half + c];
    nodes[n0 + c] += y;
  }
}

// decoder: MLP(128,256,256) via WMMA, final 256->3 via VALU dots
__global__ void k_decoder(const float* __restrict__ nodes,
                          const __bf16* w0, const float* b0,
                          const __bf16* w1, const float* b1,
                          const float* __restrict__ w2, const float* __restrict__ b2,
                          float* out) {
  __shared__ alignas(16) __bf16 IN[16 * 256];
  __shared__ alignas(16) __bf16 H1[16 * 256];
  const int lane = threadIdx.x, tile = blockIdx.x;
  const int row = lane >> 1, half = (lane & 1) * 64;
  const size_t n0 = (size_t)(tile * 16 + row) * 128 + half;
  {
    const float2* src = (const float2*)(nodes + n0);
    unsigned* dst2 = (unsigned*)&IN[row * 128 + half];
    for (int c = 0; c < 32; ++c) { float2 v = src[c]; dst2[c] = pack_bf16x2(v.x, v.y); }
  }
  __syncthreads();
  mlp_hidden(IN, 128, w0, b0, 256, H1, lane);  __syncthreads();
  mlp_hidden(H1, 256, w1, b1, 256, IN, lane);  __syncthreads();
  for (int idx = lane; idx < 48; idx += 32) {
    int m = idx / 3, c = idx % 3;
    float acc = b2[c];
    const unsigned* h2 = (const unsigned*)&IN[m * 256];
    for (int k2 = 0; k2 < 128; ++k2) {
      unsigned u = h2[k2];
      float x0 = __uint_as_float(u << 16);
      float x1 = __uint_as_float(u & 0xFFFF0000u);
      acc += x0 * w2[(2 * k2) * 3 + c] + x1 * w2[(2 * k2 + 1) * 3 + c];
    }
    out[(size_t)(tile * 16 + m) * 3 + c] = acc;
  }
}

// ---------------------------------------------------------------- host

extern "C" void kernel_launch(void* const* d_in, const int* in_sizes, int n_in,
                              void* d_out, int out_size, void* d_ws, size_t ws_size,
                              hipStream_t stream) {
  (void)in_sizes; (void)n_in; (void)out_size; (void)ws_size;
  const int*   an        = (const int*)d_in[0];
  const float* efeat     = (const float*)d_in[1];
  const int*   senders   = (const int*)d_in[2];
  const int*   receivers = (const int*)d_in[3];
  auto Fp = [&](int i) { return (const float*)d_in[i]; };

  // param leaf indices (jax pytree order: dict keys sorted, tuples/lists in order)
  // 4..9: dec W0,b0,W1,b1,W2,b2 ; 10: embedding
  // 11..18: enc_edge ln(g,be), mlp(W0,b0,W1,b1,W2,b2)
  // 19..26: enc_node ln(g,be), mlp(W0,b0,W1,b1,W2,b2)
  // layer l base L=27+20l: edge_ln g,be | edge W0..b2 | node_ln g,be | node W0..b2 |
  //                        recvW, recvb, sendW, sendb

  struct WE { int idx, K, N, Kpad; size_t off; };
  WE wes[26];
  int nw = 0;
  auto addw = [&](int idx, int K, int N, int Kpad) {
    wes[nw].idx = idx; wes[nw].K = K; wes[nw].N = N; wes[nw].Kpad = Kpad; ++nw;
  };
  addw(21, 128, 256, 128); addw(23, 256, 256, 256); addw(25, 256, 128, 256);  // enc_node
  addw(13,  23, 256,  32); addw(15, 256, 256, 256); addw(17, 256, 128, 256);  // enc_edge
  for (int l = 0; l < 3; ++l) {
    int L = 27 + 20 * l;
    addw(L + 2,  384, 256, 384); addw(L + 4,  256, 256, 256); addw(L + 6,  256, 128, 256);
    addw(L + 10, 384, 256, 384); addw(L + 12, 256, 256, 256); addw(L + 14, 256, 128, 256);
  }
  addw(4, 128, 256, 128); addw(6, 256, 256, 256);                             // decoder
  size_t welems = 0;
  for (int i = 0; i < nw; ++i) { wes[i].off = welems; welems += (size_t)wes[i].N * wes[i].Kpad; }

  // workspace carve
  size_t pos = 0;
  auto carve = [&](size_t bytes) {
    void* p = (char*)d_ws + pos;
    pos = (pos + bytes + 255) & ~(size_t)255;
    return p;
  };
  __bf16* wt    = (__bf16*)carve(welems * 2);
  float*  nodes = (float*)carve((size_t)NN * 128 * 4);
  float*  edges = (float*)carve((size_t)EE * 128 * 4);
  float*  ragg  = (float*)carve((size_t)2 * NN * 128 * 4);     // ragg|sagg contiguous
  float*  sagg  = ragg + (size_t)NN * 128;
  // ping-pong attention state: [rmax|smax|rsum|ssum] x2, [rw|sw] x2
  unsigned* statsA = (unsigned*)carve((size_t)4 * NN * 4);
  unsigned* statsB = (unsigned*)carve((size_t)4 * NN * 4);
  float* rwA = (float*)carve((size_t)EE * 4);
  float* swA = (float*)carve((size_t)EE * 4);
  float* rwB = (float*)carve((size_t)EE * 4);
  float* swB = (float*)carve((size_t)EE * 4);

  // weight conversion (f32 -> bf16 transposed)
  for (int i = 0; i < nw; ++i) {
    int total = wes[i].N * wes[i].Kpad;
    k_convert<<<(total + 255) / 256, 256, 0, stream>>>(
        Fp(wes[i].idx), wt + wes[i].off, wes[i].K, wes[i].N, wes[i].Kpad);
  }
  auto W = [&](int i) { return wt + wes[i].off; };

  // encoders (edge encoder also emits layer-0 attention logits into bank A)
  k_fill<<<(4 * NN + 255) / 256, 256, 0, stream>>>((float*)statsA, 0.0f, 4 * NN);
  k_node_encoder<<<NN / 16, 32, 0, stream>>>(an, Fp(10),
      W(0), Fp(22), W(1), Fp(24), W(2), Fp(26), Fp(19), Fp(20), nodes);
  {
    int L0 = 27;
    k_edge_encoder<<<EE / 16, 32, 0, stream>>>(efeat,
        W(3), Fp(14), W(4), Fp(16), W(5), Fp(18), Fp(11), Fp(12), edges,
        Fp(L0 + 16), Fp(L0 + 17), Fp(L0 + 18), Fp(L0 + 19),
        receivers, senders, rwA, swA, statsA, statsA + NN);
  }

  // interaction layers
  for (int l = 0; l < 3; ++l) {
    int L  = 27 + 20 * l;
    int wb = 6 + 6 * l;
    unsigned* cs = (l & 1) ? statsB : statsA;   // current stats bank
    unsigned* ns = (l & 1) ? statsA : statsB;   // next stats bank
    float* crw = (l & 1) ? rwB : rwA;
    float* csw = (l & 1) ? swB : swA;
    float* nrw = (l & 1) ? rwA : rwB;
    float* nsw = (l & 1) ? swA : swB;
    float* crsum = (float*)(cs + 2 * NN);
    float* cssum = (float*)(cs + 3 * NN);

    k_expsum<<<(EE + 255) / 256, 256, 0, stream>>>(receivers, senders,
        cs, cs + NN, crw, csw, crsum, cssum, EE);
    k_fill<<<((2 * NN * 128) + 255) / 256, 256, 0, stream>>>(ragg, 0.0f, 2 * NN * 128);

    const bool has_next = (l < 2);
    if (has_next)
      k_fill<<<(4 * NN + 255) / 256, 256, 0, stream>>>((float*)ns, 0.0f, 4 * NN);
    int Ln = 27 + 20 * (l + 1);
    k_edge_layer<<<EE / 16, 32, 0, stream>>>(edges, nodes, senders, receivers,
        W(wb + 0), Fp(L + 3), W(wb + 1), Fp(L + 5), W(wb + 2), Fp(L + 7),
        Fp(L + 0), Fp(L + 1), crw, crsum, csw, cssum, ragg, sagg,
        has_next ? Fp(Ln + 16) : nullptr, has_next ? Fp(Ln + 17) : nullptr,
        has_next ? Fp(Ln + 18) : nullptr, has_next ? Fp(Ln + 19) : nullptr,
        has_next ? nrw : nullptr, has_next ? nsw : nullptr,
        has_next ? ns : nullptr, has_next ? ns + NN : nullptr);
    k_node_layer<<<NN / 16, 32, 0, stream>>>(nodes, ragg, sagg,
        W(wb + 3), Fp(L + 11), W(wb + 4), Fp(L + 13), W(wb + 5), Fp(L + 15),
        Fp(L + 8), Fp(L + 9));
  }

  // decoder
  k_decoder<<<NN / 16, 32, 0, stream>>>(nodes,
      W(24), Fp(5), W(25), Fp(7), Fp(8), Fp(9), (float*)d_out);
}
